// MEGNetLayer_7275674599671
// MI455X (gfx1250) — matrix operand order repaired and checked by reading, not compiled, not run on hardware
//
#include <hip/hip_runtime.h>
#include <hip/hip_bf16.h>
#include <math.h>

// ---------------------------------------------------------------------------
// MEGNet layer for MI455X (gfx1250, wave32, WMMA).
// Bond/Atom MLPs run on v_wmma_f32_16x16x32_f16 tensor cores with f16
// activations/weights and f32 accumulation. All reductions are fixed-order
// (no float atomics) so replays are bit-deterministic.
//
// Round-1 fix: pass LDS tiles by array reference through __forceinline__
// helpers so clang keeps addrspace(3) provenance -> 32-bit LDS addressing,
// ds_load_b128 with immediate offsets, no generic-pointer null checks.
// ---------------------------------------------------------------------------

typedef float    v8f  __attribute__((ext_vector_type(8)));
typedef _Float16 v16h __attribute__((ext_vector_type(16)));
typedef _Float16 v8h  __attribute__((ext_vector_type(8)));
typedef _Float16 v4h  __attribute__((ext_vector_type(4)));

#define CSTR 264   // 256 + 8 pad (halfs) - comb LDS row stride
#define HSTR 136   // 128 + 8 pad (halfs) - hidden LDS row stride

// ws layout (bytes):
//   0        : Wb1t f16 [128][256]   (65536)
//   65536    : Wb2t f16 [64][128]    (16384)
//   81920    : Wa1t f16 [128][256]   (65536)
//   147456   : Wa2t f16 [64][128]    (16384)
//   163840   : bond col partials f32 [256][64]  (65536)
//   229376   : atom col partials f32 [64][64]   (16384)
// total 245760 B

__device__ __forceinline__ float softplus_f(float x) {
    return fmaxf(x, 0.0f) + log1pf(expf(-fabsf(x)));
}

__device__ __forceinline__ void cvt4h(_Float16* p, float4 v) {
    v4h h = { (_Float16)v.x, (_Float16)v.y, (_Float16)v.z, (_Float16)v.w };
    *(v4h*)p = h;
}

// A-fragment assembly from two contiguous half8 chunks
// (16-bit A 16x32, contiguous k order, same as B):
// lanes 0-15: e0..15 -> K=kc..kc+15
// lanes16-31: e0..15 -> K=kc+16..kc+31
__device__ __forceinline__ v16h combine_afrag(v8h lo, v8h hi) {
    v16h a;
#pragma unroll
    for (int i = 0; i < 8; ++i) { a[i] = lo[i]; a[i + 8] = hi[i]; }
    return a;
}

// 64-row tile MLP: out = softplus(comb @ W1 + b1) @ W2 + b2 + residual
// comb: LDS [64][CSTR] f16 (K=256), h: LDS [64][HSTR] f16 (K=128 for layer2)
// Wt1: global f16 [128][256] (transposed, n-major); Wt2: [64][128]
// Passed as array REFERENCES so addrspace(3) provenance survives inlining.
__device__ __forceinline__ void mlp_tile(
        const _Float16* __restrict__ Wt1, const float* __restrict__ b1,
        const _Float16* __restrict__ Wt2, const float* __restrict__ b2,
        _Float16 (&comb)[64][CSTR], _Float16 (&h)[64][HSTR],
        const float* __restrict__ residual, float* __restrict__ outp,
        long rowbase, long nrows) {
    const int lane = (int)(threadIdx.x & 31u);
    const int wave = (int)(threadIdx.x >> 5u);
    const int mrow = lane & 15;   // = N col for B/C frags, = M row for A frag
    const int hh   = lane >> 4;   // lane half select

    // ---- layer 1: 4 Mtiles x 8 Ntiles, K = 256 ----
    for (int t = wave; t < 32; t += 8) {
        const int m0 = (t >> 3) << 4;
        const int n0 = (t & 7) << 4;
        const float bias = b1[n0 + mrow];
        v8f acc;
#pragma unroll
        for (int r = 0; r < 8; ++r) acc[r] = bias;
        const int arow = m0 + mrow;
        const int abase = hh * 16;
        const _Float16* brow = Wt1 + (n0 + mrow) * 256 + hh * 16;
#pragma unroll
        for (int kt = 0; kt < 8; ++kt) {
            v8h lo = *(const v8h*)&comb[arow][abase + kt * 32];
            v8h hi = *(const v8h*)&comb[arow][abase + kt * 32 + 8];
            v16h a = combine_afrag(lo, hi);
            v16h b = *(const v16h*)(brow + kt * 32);
            acc = __builtin_amdgcn_wmma_f32_16x16x32_f16(false, a, false, b,
                                                         (short)0, acc, false, false);
        }
#pragma unroll
        for (int r = 0; r < 8; ++r) {
            h[m0 + hh * 8 + r][n0 + mrow] = (_Float16)softplus_f(acc[r]);
        }
    }
    __syncthreads();

    // ---- layer 2: 4 Mtiles x 4 Ntiles, K = 128 ----
    for (int t = wave; t < 16; t += 8) {
        const int m0 = (t >> 2) << 4;
        const int n0 = (t & 3) << 4;
        const float bias = b2[n0 + mrow];
        v8f acc;
#pragma unroll
        for (int r = 0; r < 8; ++r) acc[r] = bias;
        const int arow = m0 + mrow;
        const int abase = hh * 16;
        const _Float16* brow = Wt2 + (n0 + mrow) * 128 + hh * 16;
#pragma unroll
        for (int kt = 0; kt < 4; ++kt) {
            v8h lo = *(const v8h*)&h[arow][abase + kt * 32];
            v8h hi = *(const v8h*)&h[arow][abase + kt * 32 + 8];
            v16h a = combine_afrag(lo, hi);
            v16h b = *(const v16h*)(brow + kt * 32);
            acc = __builtin_amdgcn_wmma_f32_16x16x32_f16(false, a, false, b,
                                                         (short)0, acc, false, false);
        }
        const int col = n0 + mrow;
#pragma unroll
        for (int r = 0; r < 8; ++r) {
            const long grow = rowbase + m0 + hh * 8 + r;
            if (grow < nrows) {
                outp[grow * 64 + col] = acc[r] + residual[grow * 64 + col];
            }
        }
    }
}

// ---------------------------------------------------------------------------
// kernel 0: convert all MLP weights to transposed f16 in ws (once per launch)
// ---------------------------------------------------------------------------
__global__ __launch_bounds__(256) void prep_weights(
        const float* __restrict__ Wb1, const float* __restrict__ Wb2,
        const float* __restrict__ Wa1, const float* __restrict__ Wa2,
        _Float16* __restrict__ ws) {
    const int tid = (int)(blockIdx.x * 256 + threadIdx.x);
    const int stride = (int)(gridDim.x * 256);
    _Float16* wb1t = ws;            // [128][256]
    _Float16* wb2t = ws + 32768;    // [64][128]
    _Float16* wa1t = ws + 40960;    // [128][256]
    _Float16* wa2t = ws + 73728;    // [64][128]
    for (int i = tid; i < 256 * 128; i += stride) {
        int k = i >> 7, n = i & 127;
        wb1t[n * 256 + k] = (_Float16)Wb1[i];
        wa1t[n * 256 + k] = (_Float16)Wa1[i];
    }
    for (int i = tid; i < 128 * 64; i += stride) {
        int k = i >> 6, n = i & 63;
        wb2t[n * 128 + k] = (_Float16)Wb2[i];
        wa2t[n * 128 + k] = (_Float16)Wa2[i];
    }
}

// ---------------------------------------------------------------------------
// kernel 1: bond update. 64 bonds per block.
// comb_b = [atom_i | atom_j | bond | g]  (256 wide)
// ---------------------------------------------------------------------------
__global__ __launch_bounds__(256) void bond_kernel(
        const float* __restrict__ atomF, const float* __restrict__ bondF,
        const float* __restrict__ gfeat,
        const float* __restrict__ bb1, const float* __restrict__ bb2,
        const int* __restrict__ abi,
        const _Float16* __restrict__ Wt1, const _Float16* __restrict__ Wt2,
        float* __restrict__ outBonds, long nBonds) {
    __shared__ _Float16 combS[64][CSTR];
    __shared__ _Float16 hS[64][HSTR];
    const int tid = (int)threadIdx.x;
    const long rowbase = (long)blockIdx.x * 64;

    for (int it = tid; it < 1024; it += 256) {
        const int r  = it >> 4;
        const int c4 = (it & 15) << 2;
        long grow = rowbase + r;
        if (grow >= nBonds) grow = 0;
        const long ai = abi[grow * 2 + 0];
        const long aj = abi[grow * 2 + 1];
        cvt4h(&combS[r][c4],       *(const float4*)&atomF[ai * 64 + c4]);
        cvt4h(&combS[r][64 + c4],  *(const float4*)&atomF[aj * 64 + c4]);
        cvt4h(&combS[r][128 + c4], *(const float4*)&bondF[grow * 64 + c4]);
        cvt4h(&combS[r][192 + c4], *(const float4*)&gfeat[c4]);
    }
    __syncthreads();
    mlp_tile(Wt1, bb1, Wt2, bb2, combS, hS, bondF, outBonds, rowbase, nBonds);
}

// ---------------------------------------------------------------------------
// kernel 2: atom aggregation + atom update. 64 atoms per block.
// comb_a = [atom | agg | atom | g]  (256 wide)
// ---------------------------------------------------------------------------
__global__ __launch_bounds__(256) void atom_kernel(
        const float* __restrict__ atomF, const float* __restrict__ gfeat,
        const float* __restrict__ ba1, const float* __restrict__ ba2,
        const int* __restrict__ bai,
        const _Float16* __restrict__ Wt1, const _Float16* __restrict__ Wt2,
        const float* __restrict__ ubonds, float* __restrict__ outAtoms,
        long nAtoms) {
    __shared__ _Float16 combS[64][CSTR];
    __shared__ _Float16 hS[64][HSTR];
    __shared__ int   sidxS[64][32];
    __shared__ float cntS[64];
    const int tid = (int)threadIdx.x;
    const long rowbase = (long)blockIdx.x * 64;

    // stage neighbor-bond indices
    for (int i = tid; i < 64 * 32; i += 256) {
        const long grow = rowbase + (i >> 5);
        sidxS[i >> 5][i & 31] = (grow < nAtoms) ? bai[grow * 32 + (i & 31)] : -1;
    }
    __syncthreads();
    if (tid < 64) {
        int c = 0;
#pragma unroll
        for (int d = 0; d < 32; ++d) c += (sidxS[tid][d] >= 0) ? 1 : 0;
        cntS[tid] = fmaxf((float)c, 1.0f);
    }
    __syncthreads();

    for (int it = tid; it < 1024; it += 256) {
        const int r  = it >> 4;
        const int c4 = (it & 15) << 2;
        long grow = rowbase + r;
        const long g = (grow < nAtoms) ? grow : 0;
        const float4 av = *(const float4*)&atomF[g * 64 + c4];
        cvt4h(&combS[r][c4],       av);
        cvt4h(&combS[r][128 + c4], av);
        cvt4h(&combS[r][192 + c4], *(const float4*)&gfeat[c4]);
        float sx = 0.f, sy = 0.f, sz = 0.f, sw = 0.f;
        for (int d = 0; d < 32; ++d) {
            const int b = sidxS[r][d];
            if (b >= 0) {
                const float4 bv = *(const float4*)&ubonds[(long)b * 64 + c4];
                sx += bv.x; sy += bv.y; sz += bv.z; sw += bv.w;
            }
        }
        const float inv = 1.0f / cntS[r];
        cvt4h(&combS[r][64 + c4], make_float4(sx * inv, sy * inv, sz * inv, sw * inv));
    }
    __syncthreads();
    mlp_tile(Wt1, ba1, Wt2, ba2, combS, hS, atomF, outAtoms, rowbase, nAtoms);
}

// ---------------------------------------------------------------------------
// kernel 3: fixed-order per-block column partial sums (64 cols)
// ---------------------------------------------------------------------------
__global__ __launch_bounds__(256) void colsum_partial(
        const float* __restrict__ src, long nrows, float* __restrict__ partials) {
    __shared__ float red[256];
    const int tid  = (int)threadIdx.x;
    const int col  = tid & 63;
    const int roff = tid >> 6;               // 0..3
    const long step = (long)gridDim.x * 4;
    float s = 0.f;
    for (long r = (long)blockIdx.x * 4 + roff; r < nrows; r += step)
        s += src[r * 64 + col];
    red[tid] = s;
    __syncthreads();
    if (tid < 128) red[tid] += red[tid + 128];
    __syncthreads();
    if (tid < 64) partials[(long)blockIdx.x * 64 + tid] = red[tid] + red[tid + 64];
}

// ---------------------------------------------------------------------------
// kernel 4: global-state MLP (tiny) + final fixed-order reduction of partials
// comb_g = [atom_pooled | bond_pooled | g]  (192 wide)
// ---------------------------------------------------------------------------
__global__ __launch_bounds__(128) void global_update(
        const float* __restrict__ gfeat,
        const float* __restrict__ Wg1, const float* __restrict__ bg1,
        const float* __restrict__ Wg2, const float* __restrict__ bg2,
        const float* __restrict__ bondPart, const float* __restrict__ atomPart,
        float* __restrict__ outg, float invAtoms, float invBonds) {
    __shared__ float comb[192];
    __shared__ float h[128];
    const int t = (int)threadIdx.x;  // 128 threads
    if (t < 64) {
        float sa = 0.f;
        for (int b = 0; b < 64; ++b) sa += atomPart[b * 64 + t];
        comb[t] = sa * invAtoms;
        float sb = 0.f;
        for (int b = 0; b < 256; ++b) sb += bondPart[b * 64 + t];
        comb[64 + t] = sb * invBonds;
        comb[128 + t] = gfeat[t];
    }
    __syncthreads();
    float acc = bg1[t];
    for (int k = 0; k < 192; ++k) acc += comb[k] * Wg1[k * 128 + t];
    h[t] = softplus_f(acc);
    __syncthreads();
    if (t < 64) {
        float a2 = bg2[t];
        for (int k = 0; k < 128; ++k) a2 += h[k] * Wg2[k * 64 + t];
        outg[t] = a2 + gfeat[t];
    }
}

// ---------------------------------------------------------------------------
extern "C" void kernel_launch(void* const* d_in, const int* in_sizes, int n_in,
                              void* d_out, int out_size, void* d_ws, size_t ws_size,
                              hipStream_t stream) {
    const float* atomF = (const float*)d_in[0];
    const float* bondF = (const float*)d_in[1];
    const float* gF    = (const float*)d_in[2];
    const float* Wb1   = (const float*)d_in[3];
    const float* bb1   = (const float*)d_in[4];
    const float* Wb2   = (const float*)d_in[5];
    const float* bb2   = (const float*)d_in[6];
    const float* Wa1   = (const float*)d_in[7];
    const float* ba1   = (const float*)d_in[8];
    const float* Wa2   = (const float*)d_in[9];
    const float* ba2   = (const float*)d_in[10];
    const float* Wg1   = (const float*)d_in[11];
    const float* bg1   = (const float*)d_in[12];
    const float* Wg2   = (const float*)d_in[13];
    const float* bg2   = (const float*)d_in[14];
    const int*   abi   = (const int*)d_in[15];
    const int*   bai   = (const int*)d_in[16];

    const long nAtoms = (long)in_sizes[0] / 64;
    const long nBonds = (long)in_sizes[1] / 64;

    float* outAtoms  = (float*)d_out;
    float* outBonds  = outAtoms + nAtoms * 64;
    float* outGlobal = outBonds + nBonds * 64;

    _Float16* wsH      = (_Float16*)d_ws;
    _Float16* wb1t     = wsH;
    _Float16* wb2t     = wsH + 32768;
    _Float16* wa1t     = wsH + 40960;
    _Float16* wa2t     = wsH + 73728;
    float*    bondPart = (float*)((char*)d_ws + 163840);   // [256][64]
    float*    atomPart = (float*)((char*)d_ws + 229376);   // [64][64]

    prep_weights<<<128, 256, 0, stream>>>(Wb1, Wb2, Wa1, Wa2, wsH);

    const int bondBlocks = (int)((nBonds + 63) / 64);
    bond_kernel<<<bondBlocks, 256, 0, stream>>>(atomF, bondF, gF, bb1, bb2, abi,
                                                wb1t, wb2t, outBonds, nBonds);

    const int atomBlocks = (int)((nAtoms + 63) / 64);
    atom_kernel<<<atomBlocks, 256, 0, stream>>>(atomF, gF, ba1, ba2, bai,
                                                wa1t, wa2t, outBonds, outAtoms, nAtoms);

    colsum_partial<<<256, 256, 0, stream>>>(outBonds, nBonds, bondPart);
    colsum_partial<<<64, 256, 0, stream>>>(outAtoms, nAtoms, atomPart);

    global_update<<<1, 128, 0, stream>>>(gF, Wg1, bg1, Wg2, bg2, bondPart, atomPart,
                                         outGlobal, 1.0f / (float)nAtoms,
                                         1.0f / (float)nBonds);
}
